// delta_Optical_Flow_Estimator_84335977824862
// MI455X (gfx1250) — matrix operand order repaired and checked by reading, not compiled, not run on hardware
//
#include <hip/hip_runtime.h>
#include <hip/hip_bf16.h>

typedef __bf16 bf16_t;
typedef __bf16 v16bf __attribute__((ext_vector_type(16)));
typedef float  v8f   __attribute__((ext_vector_type(8)));

constexpr int HH = 192, WW = 320, BB = 8;
constexpr int HWP  = HH * WW;        // 61440
constexpr int NPIX = BB * HWP;       // 491520
constexpr int H0 = 96, W0 = 160;

__device__ __forceinline__ float leakyf(float x) { return x >= 0.f ? x : 0.1f * x; }

// ---------------- workspace layout (bytes) ----------------
constexpr size_t OFF_BF   = 0;                                  // bi_flow f32 [B,4,H,W]
constexpr size_t SZ_BF    = (size_t)4 * NPIX * 4;               // 7,864,320
constexpr size_t OFF_ACC0 = OFF_BF + SZ_BF;                     // splat acc0 f32 [B*HW, 33]
constexpr size_t SZ_ACC   = (size_t)33 * NPIX * 4;              // 64,880,640
constexpr size_t OFF_ACC1 = OFF_ACC0 + SZ_ACC;
constexpr size_t OFF_XCAT = OFF_ACC1 + SZ_ACC;                  // concat bf16 [B*HW, 160]
constexpr size_t SZ_XCAT  = (size_t)NPIX * 160 * 2;             // 157,286,400
constexpr size_t OFF_W1B  = OFF_XCAT + SZ_XCAT;                 // warped1 bf16 [B*HW, 32]
constexpr size_t SZ_W1B   = (size_t)NPIX * 32 * 2;
constexpr size_t OFF_WF   = OFF_W1B + SZ_W1B;                   // weight fragments (bf16)
// H1/H2 ping-pong reuse ACC regions (64ch bf16 = 62.9MB <= 64.9MB)

// weight fragment element offsets (bf16 elements)
constexpr size_t F1 = 0;          // 64x145: groups 4, chunks 5 -> 92160
constexpr size_t F2 = 92160;      // 64x64 : groups 4, chunks 2 -> 36864
constexpr size_t F3 = 129024;
constexpr size_t F4 = 165888;     // 32x64 : groups 2, chunks 2 -> 18432
constexpr size_t F5 = 184320;     // 32x32 : groups 2, chunks 1 -> 9216
constexpr size_t F6 = 193536;     // 16x32 : (pad 32) groups 2, chunks 1 -> 9216
constexpr size_t F7 = 202752;     // 2x16  : (pad 32x32) groups 2, chunks 1 -> 9216

// ---------------- small helper kernels ----------------
__global__ void zero_f32_kernel(float* __restrict__ p, int n) {
  int i = blockIdx.x * blockDim.x + threadIdx.x;
  if (i < n) p[i] = 0.f;
}

// bilinear 2x upsample of bi_flow_previous, * 2.0; write to BF and d_out
__global__ void upsample_kernel(const float* __restrict__ src, float* __restrict__ bf,
                                float* __restrict__ dout) {
  int o = blockIdx.x * blockDim.x + threadIdx.x;
  if (o >= 4 * NPIX) return;
  int x = o % WW;
  int y = (o / WW) % HH;
  int rest = o / (WW * HH);
  int ch = rest % 4;
  int b  = rest / 4;
  float sx = (x + 0.5f) * 0.5f - 0.5f;
  float sy = (y + 0.5f) * 0.5f - 0.5f;
  float x0f = floorf(sx), y0f = floorf(sy);
  float tx = sx - x0f, ty = sy - y0f;
  int x0 = (int)x0f, y0 = (int)y0f;
  int x1 = min(max(x0 + 1, 0), W0 - 1);
  int y1 = min(max(y0 + 1, 0), H0 - 1);
  x0 = min(max(x0, 0), W0 - 1);
  y0 = min(max(y0, 0), H0 - 1);
  const float* p = src + ((size_t)(b * 4 + ch) * H0) * W0;
  float v = (p[y0 * W0 + x0] * (1.f - tx) + p[y0 * W0 + x1] * tx) * (1.f - ty) +
            (p[y1 * W0 + x0] * (1.f - tx) + p[y1 * W0 + x1] * tx) * ty;
  v *= 2.0f;
  bf[o] = v;
  dout[o] = v;
}

// forward-splat (bilinear scatter-add) into acc[B*HW][33] (32 feat + 1 norm)
__global__ void splat_kernel(const float* __restrict__ feat, const float* __restrict__ bf,
                             int flowChOff, float* __restrict__ acc) {
  int o = blockIdx.x * blockDim.x + threadIdx.x;
  if (o >= NPIX) return;
  int b = o / HWP, idx = o % HWP;
  int y = idx / WW, x = idx % WW;
  float fx = (float)x + bf[((size_t)(b * 4 + flowChOff) * HH + y) * WW + x];
  float fy = (float)y + bf[((size_t)(b * 4 + flowChOff + 1) * HH + y) * WW + x];
  float x0 = floorf(fx), y0 = floorf(fy);
  float wx1 = fx - x0, wy1 = fy - y0;
  float wx0 = 1.f - wx1, wy0 = 1.f - wy1;
  int ix0 = (int)x0, iy0 = (int)y0;
  float v[32];
#pragma unroll
  for (int ch = 0; ch < 32; ++ch)
    v[ch] = feat[((size_t)(b * 32 + ch) * HH + y) * WW + x];
  int cx[4] = { ix0, ix0 + 1, ix0,     ix0 + 1 };
  int cy[4] = { iy0, iy0,     iy0 + 1, iy0 + 1 };
  float cw[4] = { wx0 * wy0, wx1 * wy0, wx0 * wy1, wx1 * wy1 };
  for (int k = 0; k < 4; ++k) {
    int ix = cx[k], iy = cy[k];
    if (ix < 0 || ix >= WW || iy < 0 || iy >= HH) continue;
    float wgt = cw[k];
    float* t = acc + ((size_t)b * HWP + (size_t)iy * WW + ix) * 33;
#pragma unroll
    for (int ch = 0; ch < 32; ++ch) atomicAdd(t + ch, v[ch] * wgt);
    atomicAdd(t + 32, wgt);
  }
}

// acc[.,33] -> bf16 NHWC warped feature with given pixel stride (channels 0..31)
__global__ void normalize_kernel(const float* __restrict__ acc, bf16_t* __restrict__ out,
                                 int stride) {
  int o = blockIdx.x * blockDim.x + threadIdx.x;
  if (o >= NPIX) return;
  const float* a = acc + (size_t)o * 33;
  float norm = a[32];
  float inv = 1.f / (norm == 0.f ? 1.f : norm);
  bf16_t* w = out + (size_t)o * stride;
#pragma unroll
  for (int ch = 0; ch < 32; ++ch) w[ch] = (bf16_t)(a[ch] * inv);
}

// f32 NCHW temporal feature -> bf16 into XCAT channels 32..63
__global__ void cvt_temporal_kernel(const float* __restrict__ t, bf16_t* __restrict__ xcat) {
  int o = blockIdx.x * blockDim.x + threadIdx.x;
  if (o >= NPIX) return;
  int b = o / HWP, idx = o % HWP;
  bf16_t* w = xcat + (size_t)o * 160 + 32;
#pragma unroll
  for (int ch = 0; ch < 32; ++ch)
    w[ch] = (bf16_t)t[((size_t)(b * 32 + ch) * HWP) + idx];
}

// copy warped1 (stride 32) into XCAT channels 0..31
__global__ void copy_warp1_kernel(const bf16_t* __restrict__ w1, bf16_t* __restrict__ xcat) {
  int o = blockIdx.x * blockDim.x + threadIdx.x;
  if (o >= NPIX) return;
  const bf16_t* s = w1 + (size_t)o * 32;
  bf16_t* d = xcat + (size_t)o * 160;
#pragma unroll
  for (int ch = 0; ch < 32; ++ch) d[ch] = s[ch];
}

// 9x9 correlation volume (mean over 32 ch) + leaky -> XCAT channels 64..144; zero 145..159
__global__ void corr_kernel(bf16_t* __restrict__ xcat, const bf16_t* __restrict__ w1) {
  int o = blockIdx.x * blockDim.x + threadIdx.x;
  if (o >= NPIX) return;
  int b = o / HWP, idx = o % HWP;
  int y = idx / WW, x = idx % WW;
  const bf16_t* p0 = xcat + (size_t)o * 160;
  float w0f[32];
#pragma unroll
  for (int ch = 0; ch < 32; ++ch) w0f[ch] = (float)p0[ch];
  bf16_t* vo = xcat + (size_t)o * 160 + 64;
  int i = 0;
  for (int dy = -4; dy <= 4; ++dy) {
    for (int dx = -4; dx <= 4; ++dx, ++i) {
      float s = 0.f;
      int yy = y + dy, xx = x + dx;
      if ((unsigned)yy < (unsigned)HH && (unsigned)xx < (unsigned)WW) {
        const bf16_t* p1 = w1 + ((size_t)b * HWP + (size_t)yy * WW + xx) * 32;
#pragma unroll
        for (int ch = 0; ch < 32; ++ch) s += w0f[ch] * (float)p1[ch];
      }
      vo[i] = (bf16_t)leakyf(s * (1.f / 32.f));
    }
  }
#pragma unroll
  for (int ch = 81; ch < 96; ++ch) vo[ch] = (bf16_t)0.f;  // pad 145..159
}

// Pre-swizzle OIHW f32 weights into WMMA A-fragment bf16 layout:
// flat = (((g*nChunks + c)*9 + t)*32 + lane)*16 + j ; row m = lane%16, k = 16*(lane/16) + j (contiguous k order)
__global__ void wfrag_kernel(const float* __restrict__ src, bf16_t* __restrict__ dst,
                             int Cout, int Cin, int nChunks, int total) {
  int i = blockIdx.x * blockDim.x + threadIdx.x;
  if (i >= total) return;
  int j = i & 15;
  int lane = (i >> 4) & 31;
  int rem = i >> 9;           // (g*nChunks + c)*9 + t
  int t = rem % 9;
  int gc = rem / 9;
  int c = gc % nChunks;
  int g = gc / nChunks;
  int m = g * 16 + (lane & 15);
  int kbase = (lane < 16) ? 0 : 16;
  int kk = kbase + j;
  int cin = c * 32 + kk;
  float v = 0.f;
  if (m < Cout && cin < Cin) {
    int ky = t / 3, kx = t % 3;
    v = src[((size_t)(m * Cin + cin) * 3 + ky) * 3 + kx];
  }
  dst[i] = (bf16_t)v;
}

// ---------------- WMMA implicit-GEMM 3x3 conv ----------------
// block = 128 thr (4 waves). Tile: 32 couts x 64 pixels (one image row segment).
// grid = (W/64, B*H, COUTSTRIDE/32). NHWC bf16 in/out; channel strides mult of 32.
template <int CINSTRIDE, int COUTSTRIDE, bool ACT, bool FINAL>
__global__ __launch_bounds__(128) void conv3x3_wmma(
    const bf16_t* __restrict__ in, const bf16_t* __restrict__ wfrag,
    bf16_t* __restrict__ out, float* __restrict__ fout, int foutChOff) {
  constexpr int NCH = CINSTRIDE / 32;
  __shared__ __align__(32) bf16_t tile[3][66][32];

  const int tid = threadIdx.x;
  const int lane = tid & 31;
  const int w = tid >> 5;
  const int wave_m = w >> 1;
  const int wave_n = w & 1;
  const int xb = blockIdx.x * 64;
  const int gy = blockIdx.y;
  const int b = gy / HH, y = gy % HH;
  const int g = blockIdx.z * 2 + wave_m;   // cout 16-group

  v8f acc0 = {};
  v8f acc1 = {};

  for (int c = 0; c < NCH; ++c) {
    __syncthreads();
    // cooperative LDS stage: 3 rows x 66 cols x 32 ch (this chunk), 16B units
    for (int u = tid; u < 3 * 66 * 4; u += 128) {
      int r = u / (66 * 4);
      int rem = u % (66 * 4);
      int col = rem >> 2;
      int q = rem & 3;
      int yy = y + r - 1;
      int xx = xb + col - 1;
      uint4 v = make_uint4(0u, 0u, 0u, 0u);
      if ((unsigned)yy < (unsigned)HH && (unsigned)xx < (unsigned)WW) {
        v = *(const uint4*)(in + ((size_t)((b * HH + yy) * WW + xx)) * CINSTRIDE +
                            c * 32 + q * 8);
      }
      *(uint4*)&tile[r][col][q * 8] = v;
    }
    __syncthreads();

    const int n0 = wave_n * 32 + (lane & 15);
    const int khalf = (lane >> 4) * 16;
#pragma unroll
    for (int t = 0; t < 9; ++t) {
      const int ky = t / 3, kx = t % 3;
      v16bf a = *(const v16bf*)(wfrag + ((size_t)((g * NCH + c) * 9 + t)) * 512 +
                                lane * 16);
      v16bf b0 = *(const v16bf*)&tile[ky][n0 + kx][khalf];
      acc0 = __builtin_amdgcn_wmma_f32_16x16x32_bf16(false, a, false, b0, (short)0,
                                                     acc0, false, false);
      v16bf b1 = *(const v16bf*)&tile[ky][n0 + 16 + kx][khalf];
      acc1 = __builtin_amdgcn_wmma_f32_16x16x32_bf16(false, a, false, b1, (short)0,
                                                     acc1, false, false);
    }
  }

  if constexpr (!FINAL) {
    const int coutBase = blockIdx.z * 32 + wave_m * 16 + ((lane >> 4) ? 8 : 0);
#pragma unroll
    for (int s = 0; s < 2; ++s) {
      v8f acc = (s == 0) ? acc0 : acc1;
      int n = wave_n * 32 + s * 16 + (lane & 15);
      int x = xb + n;
      union { bf16_t h[8]; uint4 u; } pk;
#pragma unroll
      for (int r = 0; r < 8; ++r) {
        float v = acc[r];
        if (ACT) v = leakyf(v);
        pk.h[r] = (bf16_t)v;
      }
      *(uint4*)(out + ((size_t)((b * HH + y) * WW + x)) * COUTSTRIDE + coutBase) = pk.u;
    }
  } else {
    // 2 real output channels; leaky then += into f32 NCHW d_out (holds bi_flow)
    if (wave_m == 0 && lane < 16 && blockIdx.z == 0) {
#pragma unroll
      for (int s = 0; s < 2; ++s) {
        v8f acc = (s == 0) ? acc0 : acc1;
        int n = wave_n * 32 + s * 16 + lane;
        int x = xb + n;
#pragma unroll
        for (int r = 0; r < 2; ++r) {
          float v = leakyf(acc[r]);
          float* p = fout + ((size_t)(b * 4 + foutChOff + r) * HH + y) * WW + x;
          *p += v;
        }
      }
    }
  }
}

// ---------------- host launch ----------------
extern "C" void kernel_launch(void* const* d_in, const int* in_sizes, int n_in,
                              void* d_out, int out_size, void* d_ws, size_t ws_size,
                              hipStream_t stream) {
  const float* feat0 = (const float*)d_in[0];
  const float* feat1 = (const float*)d_in[1];
  const float* biflow_prev = (const float*)d_in[2];
  const float* tempf = (const float*)d_in[3];
  const float* tempb = (const float*)d_in[4];
  const float* Wsrc[7] = { (const float*)d_in[5], (const float*)d_in[6],
                           (const float*)d_in[7], (const float*)d_in[8],
                           (const float*)d_in[9], (const float*)d_in[10],
                           (const float*)d_in[11] };
  float* out = (float*)d_out;
  char* ws = (char*)d_ws;

  float* BF    = (float*)(ws + OFF_BF);
  float* ACC0  = (float*)(ws + OFF_ACC0);
  float* ACC1  = (float*)(ws + OFF_ACC1);
  bf16_t* XCAT = (bf16_t*)(ws + OFF_XCAT);
  bf16_t* WARP1= (bf16_t*)(ws + OFF_W1B);
  bf16_t* WF   = (bf16_t*)(ws + OFF_WF);
  bf16_t* Hb1  = (bf16_t*)(ws + OFF_ACC0);  // ping-pong reuses acc regions
  bf16_t* Hb2  = (bf16_t*)(ws + OFF_ACC1);

  const int TPB = 256;
  const int gP = (NPIX + TPB - 1) / TPB;

  // 1) zero splat accumulators
  {
    int zn = (int)((2 * SZ_ACC) / 4);
    zero_f32_kernel<<<(zn + TPB - 1) / TPB, TPB, 0, stream>>>(ACC0, zn);
  }
  // 2) bi_flow = upsample2x(prev) * 2 -> BF and d_out
  upsample_kernel<<<(4 * NPIX + TPB - 1) / TPB, TPB, 0, stream>>>(biflow_prev, BF, out);
  // 3) forward splats
  splat_kernel<<<gP, TPB, 0, stream>>>(feat0, BF, 0, ACC0);
  splat_kernel<<<gP, TPB, 0, stream>>>(feat1, BF, 2, ACC1);
  // 4) normalize: warped0 -> XCAT[0:32], warped1 -> WARP1
  normalize_kernel<<<gP, TPB, 0, stream>>>(ACC0, XCAT, 160);
  normalize_kernel<<<gP, TPB, 0, stream>>>(ACC1, WARP1, 32);
  // 5) forward temporal -> XCAT[32:64]
  cvt_temporal_kernel<<<gP, TPB, 0, stream>>>(tempf, XCAT);
  // 6) correlation volume -> XCAT[64:145] (+zero pad 145:160)
  corr_kernel<<<gP, TPB, 0, stream>>>(XCAT, WARP1);
  // 7) weight fragments (bf16, WMMA A layout)
  {
    struct { size_t off; int cout, cin, nch, total; } L[7] = {
      { F1, 64, 145, 5, 4 * 5 * 9 * 512 },
      { F2, 64,  64, 2, 4 * 2 * 9 * 512 },
      { F3, 64,  64, 2, 4 * 2 * 9 * 512 },
      { F4, 32,  64, 2, 2 * 2 * 9 * 512 },
      { F5, 32,  32, 1, 2 * 1 * 9 * 512 },
      { F6, 16,  32, 1, 2 * 1 * 9 * 512 },
      { F7,  2,  16, 1, 2 * 1 * 9 * 512 },
    };
    for (int l = 0; l < 7; ++l) {
      wfrag_kernel<<<(L[l].total + TPB - 1) / TPB, TPB, 0, stream>>>(
          Wsrc[l], WF + L[l].off, L[l].cout, L[l].cin, L[l].nch, L[l].total);
    }
  }
  // 8) two conv stacks (forward then backward)
  dim3 blk(128);
  for (int branch = 0; branch < 2; ++branch) {
    if (branch == 1) {
      copy_warp1_kernel<<<gP, TPB, 0, stream>>>(WARP1, XCAT);   // warped1 -> XCAT[0:32]
      cvt_temporal_kernel<<<gP, TPB, 0, stream>>>(tempb, XCAT); // temporal_b -> [32:64]
      // correlation volume channels unchanged (shared)
    }
    conv3x3_wmma<160, 64, false, false><<<dim3(5, BB * HH, 2), blk, 0, stream>>>(
        XCAT, WF + F1, Hb1, nullptr, 0);
    conv3x3_wmma<64, 64, false, false><<<dim3(5, BB * HH, 2), blk, 0, stream>>>(
        Hb1, WF + F2, Hb2, nullptr, 0);
    conv3x3_wmma<64, 64, true, false><<<dim3(5, BB * HH, 2), blk, 0, stream>>>(
        Hb2, WF + F3, Hb1, nullptr, 0);   // leaky after conv3
    conv3x3_wmma<64, 32, false, false><<<dim3(5, BB * HH, 1), blk, 0, stream>>>(
        Hb1, WF + F4, Hb2, nullptr, 0);
    conv3x3_wmma<32, 32, false, false><<<dim3(5, BB * HH, 1), blk, 0, stream>>>(
        Hb2, WF + F5, Hb1, nullptr, 0);
    conv3x3_wmma<32, 32, false, false><<<dim3(5, BB * HH, 1), blk, 0, stream>>>(
        Hb1, WF + F6, Hb2, nullptr, 0);   // 16 real couts, 16 zero-pad
    conv3x3_wmma<32, 32, false, true><<<dim3(5, BB * HH, 1), blk, 0, stream>>>(
        Hb2, WF + F7, nullptr, out, branch * 2);  // leaky(delta) += bi_flow
  }
}
